// SparseMeshConv_3719441678805
// MI455X (gfx1250) — compile-verified, matrix operand order checked by reading, not run on hardware
//
#include <hip/hip_runtime.h>
#include <math.h>

typedef __attribute__((ext_vector_type(16))) __bf16 v16bf;
typedef __attribute__((ext_vector_type(8)))  __bf16 v8bf;
typedef __attribute__((ext_vector_type(4)))  __bf16 v4bf;
typedef __attribute__((ext_vector_type(8)))  float  v8f;

#define CIN     256
#define COUT    256
#define KDIM    1280     // 5 * CIN
#define NKT     40       // K tiles of 32
#define NNT     16       // N tiles of 16
#define NROWS   64       // rows per block
#define ROWPAD  1288     // bf16 per patch row: 1280 + 8 pad (16B) -> bank-conflict-free ds_load_b128
#define YSTRIDE 260      // f32 per ybuf row: 256 + 4 pad
#define PATCH_BYTES (NROWS * ROWPAD * 2)               // 164864
#define SMEM_BYTES  (PATCH_BYTES + NROWS * 16)         // + int4 idx cache = 165888

// K position of element e (0..15) within a 16-bit A/B fragment, lane-half h (0/1)
__device__ __forceinline__ int kmap(int e, int h) {
    return e + 8 * h + ((e >= 8) ? 8 : 0);
}

// ---------------------------------------------------------------------------
// Kernel 1: pack W (f32 [KDIM][COUT] row-major) -> bf16 WMMA B-fragments.
// Fragment (kt, nt): 32 lanes x 16 bf16, lane = column-in-tile (+half), so the
// GEMM kernel loads one contiguous 32B v16bf per lane.
// ---------------------------------------------------------------------------
__global__ void pack_w_kernel(const float* __restrict__ W, __bf16* __restrict__ wp) {
    int t = blockIdx.x * blockDim.x + threadIdx.x;      // one (kt, nt, lane) each
    if (t >= NKT * NNT * 32) return;
    int lane = t & 31;
    int nt   = (t >> 5) & 15;
    int kt   = t >> 9;
    int h    = lane >> 4;
    int n    = nt * 16 + (lane & 15);
    v16bf frag;
#pragma unroll
    for (int e = 0; e < 16; ++e) {
        int k = kt * 32 + kmap(e, h);
        frag[e] = (__bf16)W[k * COUT + n];
    }
    ((v16bf*)wp)[t] = frag;
}

// ---------------------------------------------------------------------------
// Kernel 2: gather+combine -> LDS patch (bf16) -> WMMA GEMM -> LN -> residual
// -> exact GELU. 256 threads = 8 waves; block covers 64 rows x 256 cols.
// ---------------------------------------------------------------------------
__global__ void __launch_bounds__(256)
mesh_conv_kernel(const float* __restrict__ x,
                 const int* __restrict__ idx1, const int* __restrict__ idx2,
                 const int* __restrict__ idx3, const int* __restrict__ idx4,
                 const __bf16* __restrict__ wp,
                 const float* __restrict__ bias,
                 const float* __restrict__ gamma, const float* __restrict__ beta,
                 float* __restrict__ out, int n)
{
    extern __shared__ unsigned char smem[];
    __bf16* patch = (__bf16*)smem;                       // [NROWS][ROWPAD] bf16
    int4*   sIdx  = (int4*)(smem + PATCH_BYTES);         // [NROWS]

    const int t       = threadIdx.x;
    const int rowBase = blockIdx.x * NROWS;

    // ---- cache gather indices for this block's rows ----
    if (t < NROWS) {
        int grow = rowBase + t;
        int4 v = {0, 0, 0, 0};
        if (grow < n) { v.x = idx1[grow]; v.y = idx2[grow]; v.z = idx3[grow]; v.w = idx4[grow]; }
        sIdx[t] = v;
    }
    __syncthreads();

    // ---- build bf16 patch rows in LDS: [x | |a-c| | a+c | |b-d| | b+d] ----
    const float4* x4 = (const float4*)x;
#pragma unroll
    for (int i = 0; i < (NROWS * 64) / 256; ++i) {       // 16 tasks/thread
        int task = i * 256 + t;
        int r    = task >> 6;                            // local row 0..63
        int q    = task & 63;                            // float4 chunk 0..63
        int grow = rowBase + r;
        v4bf seg[5];
        if (grow < n) {
            int4 id = sIdx[r];
            float4 xv = x4[grow * 64 + q];
            float4 av = x4[id.x * 64 + q];
            float4 bv = x4[id.y * 64 + q];
            float4 cv = x4[id.z * 64 + q];
            float4 dv = x4[id.w * 64 + q];
            seg[0] = (v4bf){(__bf16)xv.x, (__bf16)xv.y, (__bf16)xv.z, (__bf16)xv.w};
            seg[1] = (v4bf){(__bf16)fabsf(av.x - cv.x), (__bf16)fabsf(av.y - cv.y),
                            (__bf16)fabsf(av.z - cv.z), (__bf16)fabsf(av.w - cv.w)};
            seg[2] = (v4bf){(__bf16)(av.x + cv.x), (__bf16)(av.y + cv.y),
                            (__bf16)(av.z + cv.z), (__bf16)(av.w + cv.w)};
            seg[3] = (v4bf){(__bf16)fabsf(bv.x - dv.x), (__bf16)fabsf(bv.y - dv.y),
                            (__bf16)fabsf(bv.z - dv.z), (__bf16)fabsf(bv.w - dv.w)};
            seg[4] = (v4bf){(__bf16)(bv.x + dv.x), (__bf16)(bv.y + dv.y),
                            (__bf16)(bv.z + dv.z), (__bf16)(bv.w + dv.w)};
        } else {
            v4bf z = (v4bf){(__bf16)0.f, (__bf16)0.f, (__bf16)0.f, (__bf16)0.f};
            seg[0] = z; seg[1] = z; seg[2] = z; seg[3] = z; seg[4] = z;
        }
        __bf16* prow = patch + r * ROWPAD;
#pragma unroll
        for (int s = 0; s < 5; ++s)
            *(v4bf*)(prow + s * CIN + q * 4) = seg[s];
    }
    __syncthreads();

    // ---- WMMA GEMM: wave wv -> column tiles 2wv, 2wv+1; 4 row-groups ----
    const int wv   = t >> 5;
    const int lane = t & 31;
    const int m    = lane & 15;                           // row/col within tile
    const int h    = lane >> 4;                           // lane half
    const int nt0  = wv * 2;

    v8f acc[4][2] = {};                                   // [row-group][ntile]
    const v16bf* Wp = (const v16bf*)wp;

    for (int kt = 0; kt < NKT; ++kt) {
        v16bf B0 = Wp[(kt * NNT + nt0)     * 32 + lane];  // 32B/lane -> 2x b128
        v16bf B1 = Wp[(kt * NNT + nt0 + 1) * 32 + lane];
        const __bf16* pk = patch + kt * 32 + 8 * h;
#pragma unroll
        for (int rg = 0; rg < 4; ++rg) {
            const __bf16* arow = pk + (rg * 16 + m) * ROWPAD;
            union { v16bf v; v8bf hv[2]; } ua;
            ua.hv[0] = *(const v8bf*)(arow);              // ds_load_b128
            ua.hv[1] = *(const v8bf*)(arow + 16);         // ds_load_b128
            acc[rg][0] = __builtin_amdgcn_wmma_f32_16x16x32_bf16(
                false, ua.v, false, B0, (short)0, acc[rg][0], false, false);
            acc[rg][1] = __builtin_amdgcn_wmma_f32_16x16x32_bf16(
                false, ua.v, false, B1, (short)0, acc[rg][1], false, false);
        }
    }
    __syncthreads();                                      // done reading patch LDS

    // ---- spill acc (+bias) to f32 LDS (reuse patch region) ----
    float* ybuf = (float*)smem;                           // [NROWS][YSTRIDE]
    float bias0 = bias[nt0 * 16 + m];
    float bias1 = bias[(nt0 + 1) * 16 + m];
#pragma unroll
    for (int rg = 0; rg < 4; ++rg) {
#pragma unroll
        for (int j = 0; j < 8; ++j) {
            int r = rg * 16 + j + 8 * h;                  // C/D layout: VGPR j -> M = j (+8)
            ybuf[r * YSTRIDE + nt0 * 16 + m]       = acc[rg][0][j] + bias0;
            ybuf[r * YSTRIDE + (nt0 + 1) * 16 + m] = acc[rg][1][j] + bias1;
        }
    }
    __syncthreads();

    // ---- LayerNorm + residual + exact GELU; wave wv handles rows 8wv..8wv+7 ----
#pragma unroll 1
    for (int rr = 0; rr < 8; ++rr) {
        int r    = wv * 8 + rr;
        int grow = rowBase + r;
        if (grow >= n) continue;
        float vals[8];
        float s = 0.f, ss = 0.f;
#pragma unroll
        for (int i = 0; i < 8; ++i) {
            float v = ybuf[r * YSTRIDE + lane + 32 * i];
            vals[i] = v; s += v; ss += v * v;
        }
#pragma unroll
        for (int off = 16; off > 0; off >>= 1) {
            s  += __shfl_xor(s,  off, 32);
            ss += __shfl_xor(ss, off, 32);
        }
        float mu  = s * (1.f / 256.f);
        float var = ss * (1.f / 256.f) - mu * mu;
        float rs  = rsqrtf(var + 1e-5f);
#pragma unroll
        for (int i = 0; i < 8; ++i) {
            int col = lane + 32 * i;
            float o = (vals[i] - mu) * rs * gamma[col] + beta[col] + x[grow * CIN + col];
            o = 0.5f * o * (1.0f + erff(o * 0.70710678118654752f));
            out[grow * COUT + col] = o;
        }
    }
}

extern "C" void kernel_launch(void* const* d_in, const int* in_sizes, int n_in,
                              void* d_out, int out_size, void* d_ws, size_t ws_size,
                              hipStream_t stream) {
    const float* x     = (const float*)d_in[0];
    const int*   idx1  = (const int*)d_in[1];
    const int*   idx2  = (const int*)d_in[2];
    const int*   idx3  = (const int*)d_in[3];
    const int*   idx4  = (const int*)d_in[4];
    const float* W     = (const float*)d_in[5];
    const float* bias  = (const float*)d_in[6];
    const float* gamma = (const float*)d_in[7];
    const float* beta  = (const float*)d_in[8];
    float*       out   = (float*)d_out;
    const int    n     = in_sizes[1];                     // N = 100000

    __bf16* wp = (__bf16*)d_ws;                           // 640 KB packed bf16 W

    // deterministic, capture-safe host-side attribute set (166 KB dynamic LDS)
    (void)hipFuncSetAttribute((const void*)mesh_conv_kernel,
                              hipFuncAttributeMaxDynamicSharedMemorySize, SMEM_BYTES);

    pack_w_kernel<<<(NKT * NNT * 32 + 255) / 256, 256, 0, stream>>>(W, wp);

    int blocks = (n + NROWS - 1) / NROWS;
    mesh_conv_kernel<<<blocks, 256, SMEM_BYTES, stream>>>(
        x, idx1, idx2, idx3, idx4, wp, bias, gamma, beta, out, n);
}
